// MultiHeadAttention_15582141350369
// MI455X (gfx1250) — matrix operand order repaired and checked by reading, not compiled, not run on hardware
//
#include <hip/hip_runtime.h>
#include <cstdint>

#define D_MODEL 1024
#define NUM_HEADS 16
#define DEPTH 64
#define BATCH 4
#define SEQ 2048
#define MROWS (BATCH * SEQ)

typedef _Float16 f16x4  __attribute__((ext_vector_type(4)));
typedef _Float16 f16x8  __attribute__((ext_vector_type(8)));
typedef _Float16 f16x16 __attribute__((ext_vector_type(16)));
typedef float    f32x8  __attribute__((ext_vector_type(8)));
typedef int      i32x4  __attribute__((ext_vector_type(4)));
typedef int      i32x8  __attribute__((ext_vector_type(8)));

static __device__ __forceinline__ f16x16 cat8(f16x8 lo, f16x8 hi) {
  return __builtin_shufflevector(lo, hi, 0,1,2,3,4,5,6,7,8,9,10,11,12,13,14,15);
}

// A-fragment (16x32 f16): lane = (m = lane&15, half = lane>>4).
// e<8 -> K = half*8 + e ; e>=8 -> K = 16 + half*8 + (e-8). Two 16B LDS loads.
static __device__ __forceinline__ f16x16 ld_afrag(const _Float16* rowp, int lh) {
  f16x8 lo = *(const f16x8*)(rowp + lh * 8);
  f16x8 hi = *(const f16x8*)(rowp + 16 + lh * 8);
  return cat8(lo, hi);
}

// B-fragment (32x16 f16), storage transposed [n][k]: e<8 -> K = half*8+e ; e>=8 -> K = 16+half*8+(e-8).
static __device__ __forceinline__ f16x16 ld_bfrag(const _Float16* rowp, int lh) {
  f16x8 lo = *(const f16x8*)(rowp + lh * 8);
  f16x8 hi = *(const f16x8*)(rowp + 16 + lh * 8);
  return cat8(lo, hi);
}

static __device__ __forceinline__ f32x8 wmma_f16(f16x16 a, f16x16 b, f32x8 c) {
  return __builtin_amdgcn_wmma_f32_16x16x32_f16(false, a, false, b, (short)0, c,
                                                false, false);
}

// ---------------------------------------------------------------------------
// Tensor Data Mover: async 2D f16 tile load Global -> LDS.
// pad codes: interval c -> 2^(c+1) dwords between pads; amount c -> c+1 dwords.
// Issued wave-uniform; completion via s_wait_tensorcnt (TENSORcnt).
// ---------------------------------------------------------------------------
static __device__ __forceinline__ void tdm_load_2d_f16(
    unsigned lds_off, const void* gaddr, unsigned tile_d0, unsigned tile_d1,
    unsigned tensor_d0, unsigned tensor_d1, unsigned stride_d0,
    unsigned pad_int_code, unsigned pad_amt_code) {
  const unsigned long long ga = (unsigned long long)(uintptr_t)gaddr;
  i32x4 g0;
  g0[0] = 1;  // count=1 (valid user descriptor)
  g0[1] = (int)lds_off;
  g0[2] = (int)(unsigned)(ga & 0xffffffffull);
  g0[3] = (int)((unsigned)((ga >> 32) & 0x1ffffffull) | (2u << 30));
  i32x8 g1;
  g1[0] = (int)((1u << 16) | (1u << 20) | (pad_int_code << 22) |
                (pad_amt_code << 25));
  g1[1] = (int)((tensor_d0 & 0xffffu) << 16);
  g1[2] = (int)(((tensor_d0 >> 16) & 0xffffu) | ((tensor_d1 & 0xffffu) << 16));
  g1[3] = (int)(((tensor_d1 >> 16) & 0xffffu) | ((tile_d0 & 0xffffu) << 16));
  g1[4] = (int)(tile_d1 & 0xffffu);
  g1[5] = (int)stride_d0;
  g1[6] = 0;
  g1[7] = 0;
  asm volatile("tensor_load_to_lds %0, %1" ::"s"(g0), "s"(g1) : "memory");
}

// ---------------------------------------------------------------------------
// Weight prep: Wt[n][k] = (f16) W[k][n]  (1024x1024), LDS tile transpose.
// ---------------------------------------------------------------------------
__global__ __launch_bounds__(256) void transpose_convert_kernel(
    const float* __restrict__ W, _Float16* __restrict__ Wt) {
  __shared__ float tile[32][33];
  const int tx = threadIdx.x & 31, ty = threadIdx.x >> 5;
  const int k0 = blockIdx.x * 32, n0 = blockIdx.y * 32;
#pragma unroll
  for (int i = 0; i < 32; i += 8)
    tile[ty + i][tx] = W[(size_t)(k0 + ty + i) * D_MODEL + n0 + tx];
  __syncthreads();
#pragma unroll
  for (int i = 0; i < 32; i += 8)
    Wt[(size_t)(n0 + ty + i) * D_MODEL + k0 + tx] = (_Float16)tile[tx][ty + i];
}

// ---------------------------------------------------------------------------
// GEMM: C[M,N] = A[M,K] * Bt^T + bias, Bt is pre-transposed f16 [N][K].
// Block = 256 thr = 8 waves, tile 128x128, K-chunk 32; wave -> 32x64 (2x4 WMMA).
// Double-buffered LDS: TDM for chunk i+1 issued before computing chunk i.
// OMODE: 0 = f16 row-major, 1 = f32 row-major (d_out), 2 = f16 [b,h,d,s].
// ---------------------------------------------------------------------------
template <bool A_IS_F32, int OMODE>
__global__ __launch_bounds__(256) void gemm_wmma_kernel(
    const void* __restrict__ Av, const _Float16* __restrict__ Bt,
    const float* __restrict__ bias, void* __restrict__ Cout, int M, int N,
    int K) {
  // pool: buf0 {As@0, Bs@10240B}, buf1 {As@20480B, Bs@30720B}
  alignas(16) __shared__ _Float16 lds_pool[4 * 128 * 40];
  const int tid = threadIdx.x;
  const int lane = tid & 31;
  const int l16 = lane & 15;
  const int lh = (lane >> 4) & 1;
  const int w = tid >> 5;
  const int wm = w >> 1;  // 0..3
  const int wn = w & 1;   // 0..1
  const int m0 = blockIdx.x * 128;
  const int n0 = blockIdx.y * 128;

  auto issue_tdm = [&](int buf, int k0) {
    // B tile: Bt rows n0..n0+127, cols k0..k0+31; pad 16dw+4dw -> stride 40h
    tdm_load_2d_f16((unsigned)(buf * 20480 + 10240), Bt + (size_t)n0 * K + k0,
                    32, 128, (unsigned)K, (unsigned)N, (unsigned)K, 3, 3);
    if constexpr (!A_IS_F32)
      tdm_load_2d_f16((unsigned)(buf * 20480),
                      (const _Float16*)Av + (size_t)m0 * K + k0, 32, 128,
                      (unsigned)K, (unsigned)M, (unsigned)K, 3, 3);
  };
  auto stage_a_f32 = [&](int buf, int k0) {
    _Float16* As = lds_pool + buf * (2 * 128 * 40);
#pragma unroll
    for (int i = 0; i < 4; ++i) {
      int q = tid + i * 256;  // quad id; 8 quads per row of 32
      int r = q >> 3;
      int c = (q & 7) * 4;
      const float4 v =
          *(const float4*)((const float*)Av + (size_t)(m0 + r) * K + k0 + c);
      As[r * 40 + c + 0] = (_Float16)v.x;
      As[r * 40 + c + 1] = (_Float16)v.y;
      As[r * 40 + c + 2] = (_Float16)v.z;
      As[r * 40 + c + 3] = (_Float16)v.w;
    }
  };

  f32x8 acc[2][4] = {};

  // prologue: fill buffer 0
  if (w == 0) issue_tdm(0, 0);
  if constexpr (A_IS_F32) stage_a_f32(0, 0);
  if (w == 0) __builtin_amdgcn_s_wait_tensorcnt(0);
  __syncthreads();

  for (int k0 = 0; k0 < K; k0 += 32) {
    const int cur = (k0 >> 5) & 1;
    const int nxt = cur ^ 1;
    if (k0 + 32 < K) {  // kick off next chunk's DMA / staging
      if (w == 0) issue_tdm(nxt, k0 + 32);
      if constexpr (A_IS_F32) {
        stage_a_f32(nxt, k0 + 32);
        if (k0 + 64 < K)
          __builtin_prefetch(
              (const float*)Av + (size_t)(m0 + (tid >> 1)) * K + k0 + 64, 0, 1);
      }
    }
    const _Float16* As = lds_pool + cur * (2 * 128 * 40);
    const _Float16* Bs = As + 128 * 40;
    f16x16 af[2], bf[4];
#pragma unroll
    for (int mt = 0; mt < 2; ++mt)
      af[mt] = ld_afrag(&As[(wm * 32 + mt * 16 + l16) * 40], lh);
#pragma unroll
    for (int nt = 0; nt < 4; ++nt)
      bf[nt] = ld_bfrag(&Bs[(wn * 64 + nt * 16 + l16) * 40], lh);
#pragma unroll
    for (int mt = 0; mt < 2; ++mt)
#pragma unroll
      for (int nt = 0; nt < 4; ++nt)
        acc[mt][nt] = wmma_f16(af[mt], bf[nt], acc[mt][nt]);

    if (w == 0) __builtin_amdgcn_s_wait_tensorcnt(0);
    __syncthreads();
  }

#pragma unroll
  for (int mt = 0; mt < 2; ++mt) {
#pragma unroll
    for (int nt = 0; nt < 4; ++nt) {
      const int col = n0 + wn * 64 + nt * 16 + l16;
      const float bv = bias[col];
      if constexpr (OMODE == 2) {
        // f16, head-transposed [b, h, d, s]; 8 consecutive s -> one 16B store
        const int hh = col >> 6, d = col & 63;
        const int row0 = m0 + wm * 32 + mt * 16 + 8 * lh;
        const int bb = row0 >> 11, s = row0 & (SEQ - 1);
        f16x8 vv;
#pragma unroll
        for (int r = 0; r < 8; ++r) vv[r] = (_Float16)(acc[mt][nt][r] + bv);
        *(f16x8*)((_Float16*)Cout +
                  ((size_t)(bb * NUM_HEADS + hh) * DEPTH + d) * SEQ + s) = vv;
      } else {
#pragma unroll
        for (int r = 0; r < 8; ++r) {
          const int row = m0 + wm * 32 + mt * 16 + r + 8 * lh;
          const float v = acc[mt][nt][r] + bv;
          if constexpr (OMODE == 1)
            ((float*)Cout)[(size_t)row * N + col] = v;
          else
            ((_Float16*)Cout)[(size_t)row * N + col] = (_Float16)v;
        }
      }
    }
  }
}

// ---------------------------------------------------------------------------
// Flash attention: block = (64 q-rows, head, batch), 4 waves x 16 q-rows.
// K-chunk = 64, double-buffered K/V TDM tiles.  V comes from head-transposed
// [b,h,d,s] so its tile lands directly in B-fragment layout.  Row-sum of P is
// a ones-column appended to V (5th WMMA N-tile) -> no sum shuffles.
// ---------------------------------------------------------------------------
__global__ __launch_bounds__(128) void flash_attn_kernel(
    const _Float16* __restrict__ Qf, const _Float16* __restrict__ Kf,
    const _Float16* __restrict__ Vt, const unsigned char* __restrict__ mask,
    _Float16* __restrict__ Of) {
  // LDS pool (halfs / byte offsets); buffer stride = 10368 halfs (20736 B):
  //   Qs  [64][72] @ 0
  //   Ks0 [64][72] @ 9216 B   Vs0 [80][72] @ 18432 B
  //   Ks1 [64][72] @ 29952 B  Vs1 [80][72] @ 39168 B
  //   Ps 4x[16][72] @ 50688 B                  (total 59904 B)
  alignas(16) __shared__ _Float16 lds_pool[29952];
  _Float16* Qs = lds_pool;
  _Float16* Ps = lds_pool + 25344;
#define PS_(wv, r, c) Ps[((wv) * 16 + (r)) * 72 + (c)]

  const int tid = threadIdx.x;
  const int lane = tid & 31;
  const int l16 = lane & 15;
  const int lh = (lane >> 4) & 1;
  const int w = tid >> 5;  // 0..3
  const int h = blockIdx.y, b = blockIdx.z;
  const int q0 = blockIdx.x * 64;
  const size_t headoff = (size_t)h * DEPTH;

  auto issue_kv = [&](int buf, int kb) {
    tdm_load_2d_f16(9216u + (unsigned)buf * 20736u,
                    Kf + (size_t)(b * SEQ + kb) * D_MODEL + headoff, 64, 64,
                    D_MODEL, SEQ, D_MODEL, 4, 3);
    tdm_load_2d_f16(18432u + (unsigned)buf * 20736u,
                    Vt + (size_t)(b * NUM_HEADS + h) * DEPTH * SEQ + kb, 64, 64,
                    SEQ, DEPTH, SEQ, 4, 3);
  };

  // ones/zeros rows 64..79 of both V buffers (row 64 = row-sum column)
  for (int i = tid; i < 16 * 72; i += 128) {
    int rr = i / 72, cc = i % 72;
    _Float16 v = (rr == 0) ? (_Float16)1.0f : (_Float16)0.0f;
    lds_pool[9216 + (64 + rr) * 72 + cc] = v;           // Vs0 @ 9216 halfs
    lds_pool[9216 + 10368 + (64 + rr) * 72 + cc] = v;   // Vs1 @ 19584 halfs
  }
  if (w == 0) {  // Q tile + first K/V chunk
    tdm_load_2d_f16(0, Qf + (size_t)(b * SEQ + q0) * D_MODEL + headoff, 64, 64,
                    D_MODEL, SEQ, D_MODEL, 4, 3);
    issue_kv(0, 0);
    __builtin_amdgcn_s_wait_tensorcnt(0);
  }
  __syncthreads();

  float m_i[8];
#pragma unroll
  for (int r = 0; r < 8; ++r) m_i[r] = -1e30f;
  f32x8 oacc[5] = {};
  const float scale = 0.125f;  // 1/sqrt(DEPTH)

  for (int kb = 0; kb < SEQ; kb += 64) {
    const int cur = (kb >> 6) & 1;
    if (kb + 64 < SEQ && w == 0) issue_kv(cur ^ 1, kb + 64);
    const _Float16* Ks = lds_pool + 4608 + cur * 10368;  // runtime GEP, no
    const _Float16* Vs = lds_pool + 9216 + cur * 10368;  // static ptr tables

    // ---- S = Q K^T : 16 q-rows x 64 k-cols, depth 64 (2 K-steps) ----
    f16x16 aq0 = ld_afrag(&Qs[(w * 16 + l16) * 72], lh);
    f16x16 aq1 = ld_afrag(&Qs[(w * 16 + l16) * 72 + 32], lh);
    f32x8 s4[4] = {};
#pragma unroll
    for (int nt = 0; nt < 4; ++nt) {
      f16x16 bk0 = ld_bfrag(&Ks[(nt * 16 + l16) * 72], lh);
      f16x16 bk1 = ld_bfrag(&Ks[(nt * 16 + l16) * 72 + 32], lh);
      s4[nt] = wmma_f16(aq0, bk0, s4[nt]);
      s4[nt] = wmma_f16(aq1, bk1, s4[nt]);
    }

    // ---- scale + key-padding mask (lane = one k column per tile) ----
    float sv[4][8];
#pragma unroll
    for (int nt = 0; nt < 4; ++nt) {
      const int kcol = kb + nt * 16 + l16;
      const bool valid = mask[(size_t)b * SEQ + kcol] != 0;
#pragma unroll
      for (int r = 0; r < 8; ++r)
        sv[nt][r] = valid ? s4[nt][r] * scale : -1e30f;
    }

    // ---- row max over 64 cols (in-lane over tiles, then 4 xor-shuffles) ----
    float mnew[8], alpha[8];
#pragma unroll
    for (int r = 0; r < 8; ++r) {
      float mx = fmaxf(fmaxf(sv[0][r], sv[1][r]), fmaxf(sv[2][r], sv[3][r]));
#pragma unroll
      for (int off = 8; off >= 1; off >>= 1)
        mx = fmaxf(mx, __shfl_xor(mx, off, 32));
      mnew[r] = fmaxf(m_i[r], mx);
      alpha[r] = __expf(m_i[r] - mnew[r]);
      m_i[r] = mnew[r];
    }
    // ---- P = exp(S - m)  (C-layout -> A-layout transpose via LDS) ----
#pragma unroll
    for (int nt = 0; nt < 4; ++nt)
#pragma unroll
      for (int r = 0; r < 8; ++r)
        PS_(w, r + 8 * lh, nt * 16 + l16) = (_Float16)__expf(sv[nt][r] - mnew[r]);
    // ---- rescale O (and running-sum column) ----
#pragma unroll
    for (int nt = 0; nt < 5; ++nt)
#pragma unroll
      for (int r = 0; r < 8; ++r) oacc[nt][r] *= alpha[r];

    __syncthreads();  // publish Ps before A-fragment reads

    // ---- O += P V  (K = 64 k-positions, N = 64 depth + 16 "ones" cols) ----
    f16x16 ap0 = ld_afrag(&PS_(w, l16, 0), lh);
    f16x16 ap1 = ld_afrag(&PS_(w, l16, 32), lh);
#pragma unroll
    for (int nt = 0; nt < 5; ++nt) {
      f16x16 bv0 = ld_bfrag(&Vs[(nt * 16 + l16) * 72], lh);
      f16x16 bv1 = ld_bfrag(&Vs[(nt * 16 + l16) * 72 + 32], lh);
      oacc[nt] = wmma_f16(ap0, bv0, oacc[nt]);
      oacc[nt] = wmma_f16(ap1, bv1, oacc[nt]);
    }

    if (w == 0) __builtin_amdgcn_s_wait_tensorcnt(0);
    __syncthreads();  // cur buffers free for next DMA; nxt buffers ready
  }

  // ---- l lives in tile 4 (col 64 -> lane l16==0); broadcast + normalize ----
  float inv[8];
#pragma unroll
  for (int r = 0; r < 8; ++r) {
    const float lsum = __shfl(oacc[4][r], lh * 16, 32);
    inv[r] = lsum > 0.0f ? 1.0f / lsum : 0.0f;
  }
#pragma unroll
  for (int nt = 0; nt < 4; ++nt) {
    const int d = nt * 16 + l16;
#pragma unroll
    for (int r = 0; r < 8; ++r) {
      const int q = q0 + w * 16 + r + 8 * lh;
      Of[(size_t)(b * SEQ + q) * D_MODEL + headoff + d] =
          (_Float16)(oacc[nt][r] * inv[r]);
    }
  }
#undef PS_
}

// ---------------------------------------------------------------------------
extern "C" void kernel_launch(void* const* d_in, const int* in_sizes, int n_in,
                              void* d_out, int out_size, void* d_ws,
                              size_t ws_size, hipStream_t stream) {
  (void)in_sizes; (void)n_in; (void)out_size; (void)ws_size;
  const float* value = (const float*)d_in[0];
  const float* key   = (const float*)d_in[1];
  const float* query = (const float*)d_in[2];
  const unsigned char* mask = (const unsigned char*)d_in[3];
  const float* Wq = (const float*)d_in[4];
  const float* bq = (const float*)d_in[5];
  const float* Wk = (const float*)d_in[6];
  const float* bk = (const float*)d_in[7];
  const float* Wv = (const float*)d_in[8];
  const float* bv = (const float*)d_in[9];
  const float* Wo = (const float*)d_in[10];
  const float* bo = (const float*)d_in[11];

  const size_t ACT = (size_t)MROWS * D_MODEL;   // 8M halfs
  const size_t WEL = (size_t)D_MODEL * D_MODEL; // 1M halfs
  _Float16* Qh  = (_Float16*)d_ws;        // row-major [b*s][1024]
  _Float16* Kh  = Qh + ACT;
  _Float16* VhT = Kh + ACT;               // head-transposed [b][h][d][s]
  _Float16* Ah  = VhT + ACT;              // attention output, row-major
  _Float16* Wqt = Ah + ACT;               // transposed f16 weights [n][k]
  _Float16* Wkt = Wqt + WEL;
  _Float16* Wvt = Wkt + WEL;
  _Float16* Wot = Wvt + WEL;              // total ws: 72 MB

  dim3 gT(D_MODEL / 32, D_MODEL / 32, 1);
  transpose_convert_kernel<<<gT, 256, 0, stream>>>(Wq, Wqt);
  transpose_convert_kernel<<<gT, 256, 0, stream>>>(Wk, Wkt);
  transpose_convert_kernel<<<gT, 256, 0, stream>>>(Wv, Wvt);
  transpose_convert_kernel<<<gT, 256, 0, stream>>>(Wo, Wot);

  dim3 gG(MROWS / 128, D_MODEL / 128, 1);
  gemm_wmma_kernel<true, 0><<<gG, 256, 0, stream>>>(query, Wqt, bq, Qh, MROWS,
                                                    D_MODEL, D_MODEL);
  gemm_wmma_kernel<true, 0><<<gG, 256, 0, stream>>>(key, Wkt, bk, Kh, MROWS,
                                                    D_MODEL, D_MODEL);
  gemm_wmma_kernel<true, 2><<<gG, 256, 0, stream>>>(value, Wvt, bv, VhT, MROWS,
                                                    D_MODEL, D_MODEL);

  dim3 gF(SEQ / 64, NUM_HEADS, BATCH);
  flash_attn_kernel<<<gF, 128, 0, stream>>>(Qh, Kh, VhT, mask, Ah);

  gemm_wmma_kernel<false, 1><<<gG, 256, 0, stream>>>(Ah, Wot, bo, d_out, MROWS,
                                                     D_MODEL, D_MODEL);
}
